// SegmentationWithAttention_23656679867646
// MI455X (gfx1250) — matrix operand order repaired and checked by reading, not compiled, not run on hardware
//
#include <hip/hip_runtime.h>
#include <hip/hip_bf16.h>

// Shapes from the reference
#define Bn   4
#define Sn   256
#define En   256
#define Hn   8
#define DHn  32
#define LDn  128
#define Pn   256
#define Nn   (Sn * Pn)      // 65536
#define NCn  2
#define C2n  128            // E/2
#define C3n  64
#define EPSf 1e-5f

typedef float v2f __attribute__((ext_vector_type(2)));
typedef float v8f __attribute__((ext_vector_type(8)));
typedef _Float16 v8h  __attribute__((ext_vector_type(8)));
typedef _Float16 v16h __attribute__((ext_vector_type(16)));

__device__ __forceinline__ v8f wmma_f32_4(v2f a, v2f b, v8f c) {
    // V_WMMA_F32_16X16X4_F32 : D = A(16x4) * B(4x16) + C, full fp32
    return __builtin_amdgcn_wmma_f32_16x16x4_f32(false, a, false, b, (short)0, c,
                                                 false, false);
}
__device__ __forceinline__ v8f wmma_f16_32(v16h a, v16h b, v8f c) {
    // V_WMMA_F32_16X16X32_F16 : D = A(16x32) * B(32x16) + C, f32 accumulate
    return __builtin_amdgcn_wmma_f32_16x16x32_f16(false, a, false, b, (short)0, c,
                                                  false, false);
}

// ---------------------------------------------------------------------------
// Kernel 1: x[b,s,e] = gl_feats[s,b,e] + pos(b,s,e)
// ---------------------------------------------------------------------------
__global__ void pos_x_kernel(const float* __restrict__ gl,
                             const float* __restrict__ cent,
                             const float* __restrict__ fc1w,
                             const float* __restrict__ fc1b,
                             const float* __restrict__ fc2w,
                             const float* __restrict__ fc2b,
                             float* __restrict__ x) {
    int idx = blockIdx.x * blockDim.x + threadIdx.x;
    if (idx >= Bn * Sn * En) return;
    int e  = idx & (En - 1);
    int bs = idx >> 8;               // b*256 + s
    int s  = bs & (Sn - 1);
    int b  = bs >> 8;
    float c0 = cent[bs * 2 + 0];
    float c1 = cent[bs * 2 + 1];
    float p = fc2b[e];
#pragma unroll
    for (int j = 0; j < 16; ++j) {
        float h = c0 * fc1w[j * 2 + 0] + c1 * fc1w[j * 2 + 1] + fc1b[j];
        h = (h >= 0.f) ? h : 0.01f * h;
        p += h * fc2w[e * 16 + j];
    }
    x[idx] = gl[(s * Bn + b) * En + e] + p;
}

// ---------------------------------------------------------------------------
// Weight pre-conversion: f32 -> f16 for the hot fused-MLP weights.
// ---------------------------------------------------------------------------
__global__ void convert_weights_kernel(const float* __restrict__ w2,
                                       const float* __restrict__ w3,
                                       _Float16* __restrict__ w2h,
                                       _Float16* __restrict__ w3h) {
    int i = blockIdx.x * blockDim.x + threadIdx.x;
    if (i < C2n * LDn) {
        int n = i >> 7, k = i & 127;
        w2h[i] = (_Float16)w2[n * 384 + k];
    }
    if (i < C3n * C2n) {
        w3h[i] = (_Float16)w3[i];
    }
}

// ---------------------------------------------------------------------------
// Generic fp32 WMMA GEMM: C[M,N] = A[M,K](lda) @ W[N,K](ldw)^T + bias
// One wave per 16x16 output tile.  (kept fp32: feeds softmax / g2 path)
// ---------------------------------------------------------------------------
__global__ void gemm_wmma_f32(const float* __restrict__ A, int lda,
                              const float* __restrict__ W, int ldw,
                              const float* __restrict__ bias,
                              float* __restrict__ C, int ldc,
                              int M, int N, int K) {
    int wave = (blockIdx.x * blockDim.x + threadIdx.x) >> 5;
    int lane = threadIdx.x & 31;
    int ntiles = N >> 4;
    int mt = wave / ntiles;
    int nt = wave - mt * ntiles;
    if (mt * 16 >= M) return;                 // wave-uniform
    int half = lane >> 4;
    int l16  = lane & 15;

    const float* Arow = A + (mt * 16 + l16) * lda;
    const float* Wrow = W + (nt * 16 + l16) * ldw;
    v8f acc = {};
    for (int k0 = 0; k0 < K; k0 += 4) {
        int ka = k0 + 2 * half;
        v2f a; a.x = Arow[ka]; a.y = Arow[ka + 1];
        v2f b; b.x = Wrow[ka]; b.y = Wrow[ka + 1];
        acc = wmma_f32_4(a, b, acc);
    }
    float bv = bias ? bias[nt * 16 + l16] : 0.f;
    int col = nt * 16 + l16;
#pragma unroll
    for (int i = 0; i < 8; ++i) {
        int row = mt * 16 + i + 8 * half;
        C[row * ldc + col] = acc[i] + bv;
    }
}

// ---------------------------------------------------------------------------
// Kernel 3: attention. One wave per (b, h, q-row). Streaming softmax, fp32.
// ---------------------------------------------------------------------------
__global__ void attention_kernel(const float* __restrict__ qkv,
                                 float* __restrict__ ctx) {
    int wid  = (blockIdx.x * blockDim.x + threadIdx.x) >> 5;
    int lane = threadIdx.x & 31;
    int q = wid & (Sn - 1);
    int h = (wid >> 8) & (Hn - 1);
    int b = wid >> 11;

    const float* base = qkv + (size_t)(b * Sn) * (3 * En);
    const float* qrow = base + q * (3 * En) + h * DHn;

    float qv[DHn];
#pragma unroll
    for (int d = 0; d < DHn; ++d) qv[d] = qrow[d];

    const float scale = 0.1767766952966369f;   // 1/sqrt(32)
    float sc[8];
#pragma unroll
    for (int t = 0; t < 8; ++t) {
        int k = t * 32 + lane;
        const float* krow = base + k * (3 * En) + En + h * DHn;
        float s = 0.f;
#pragma unroll
        for (int d = 0; d < DHn; ++d) s += qv[d] * krow[d];
        sc[t] = s * scale;
    }
    float m = sc[0];
#pragma unroll
    for (int t = 1; t < 8; ++t) m = fmaxf(m, sc[t]);
#pragma unroll
    for (int off = 16; off >= 1; off >>= 1) m = fmaxf(m, __shfl_xor(m, off, 32));
    float lsum = 0.f;
#pragma unroll
    for (int t = 0; t < 8; ++t) { sc[t] = __expf(sc[t] - m); lsum += sc[t]; }
#pragma unroll
    for (int off = 16; off >= 1; off >>= 1) lsum += __shfl_xor(lsum, off, 32);
    float inv = 1.f / lsum;

    float acc = 0.f;
#pragma unroll
    for (int t = 0; t < 8; ++t) {
        for (int j = 0; j < 32; ++j) {
            float p = __shfl(sc[t], j, 32);
            int k = t * 32 + j;
            acc += p * base[k * (3 * En) + 2 * En + h * DHn + lane];
        }
    }
    ctx[(b * Sn + q) * En + h * DHn + lane] = acc * inv;
}

// ---------------------------------------------------------------------------
// Kernel 6: fused per-point MLP with f16 WMMA (f32 accumulate).
// 32 rows per wave (2 m-tiles share each B fetch), 4 waves/block -> 128 rows.
// K-chunk loops fully unrolled: 64 static WMMA (conv2) + 32 (conv3) per wave.
// ---------------------------------------------------------------------------
#define SH_H16  136   // f16 row stride (spreads LDS banks)
#define SH_F32  66    // f32 row stride for h3 tile

__global__ void fused_mlp_kernel(const float* __restrict__ lo,
                                 const float* __restrict__ g2,
                                 const _Float16* __restrict__ w2h,  // (128,128) f16
                                 const float* __restrict__ bn2g, const float* __restrict__ bn2b,
                                 const float* __restrict__ bn2m, const float* __restrict__ bn2v,
                                 const _Float16* __restrict__ w3h,  // (64,128) f16
                                 const float* __restrict__ b3,
                                 const float* __restrict__ bn3g, const float* __restrict__ bn3b,
                                 const float* __restrict__ bn3m, const float* __restrict__ bn3v,
                                 const float* __restrict__ w4,      // (2,64) f32
                                 const float* __restrict__ b4,
                                 float* __restrict__ out) {
    __shared__ _Float16 shh[4 * 2 * 16 * SH_H16];  // h2 tiles (f16)  ~34.8 KB
    __shared__ float    shf[4 * 2 * 16 * SH_F32];  // h3 tiles (f32)  ~33.8 KB

    int waveId = threadIdx.x >> 5;
    int lane   = threadIdx.x & 31;
    int half   = lane >> 4;
    int l16    = lane & 15;

    int waveTile = blockIdx.x * 4 + waveId;   // 32-row super-tile index
    int r0 = waveTile * 32;                   // flat row in [0, B*N)
    int bIdx = r0 >> 16;
    int n0   = r0 & (Nn - 1);
    int s    = n0 >> 8;                       // cluster id (constant: 32 | 256)

    // ---- conv2 (lo half): 2x (16x128) tiles, K=128 as 4 chunks of 32 ----
    const float* arow0 = lo + (size_t)(r0 + l16) * LDn;        // m-tile 0
    const float* arow1 = lo + (size_t)(r0 + 16 + l16) * LDn;   // m-tile 1
    v8f acc2[2][8] = {};
#pragma unroll
    for (int c = 0; c < 4; ++c) {
        int kb = c * 32;
        int b0 = kb + 16 * half;       // A VGPRs 0..3 : K = b0 .. b0+7
        int b1 = kb + 16 * half + 8;   // A VGPRs 4..7 : K = b1 .. b1+7
        v8f f00 = *(const v8f*)(arow0 + b0);   // 32B aligned
        v8f f01 = *(const v8f*)(arow0 + b1);
        v8f f10 = *(const v8f*)(arow1 + b0);
        v8f f11 = *(const v8f*)(arow1 + b1);
        v16h a0, a1;
#pragma unroll
        for (int j = 0; j < 8; ++j) {
            a0[j] = (_Float16)f00[j];  a0[8 + j] = (_Float16)f01[j];
            a1[j] = (_Float16)f10[j];  a1[8 + j] = (_Float16)f11[j];
        }
        int kbB = kb + 16 * half;      // B: lane l16 = col, K contiguous
#pragma unroll
        for (int nt = 0; nt < 8; ++nt) {
            v16h bb = *(const v16h*)(w2h + (nt * 16 + l16) * LDn + kbB);
            acc2[0][nt] = wmma_f16_32(a0, bb, acc2[0][nt]);
            acc2[1][nt] = wmma_f16_32(a1, bb, acc2[1][nt]);
        }
    }
    // g2 (global part + conv2_b) + bn2 + relu -> stash h2 tiles as f16 in LDS
    _Float16* shw = shh + waveId * (2 * 16 * SH_H16);
    const float* g2row = g2 + (size_t)(bIdx * Sn + s) * C2n;
#pragma unroll
    for (int nt = 0; nt < 8; ++nt) {
        int cc = nt * 16 + l16;
        float gadd  = g2row[cc];
        float scl   = bn2g[cc] * rsqrtf(bn2v[cc] + EPSf);
        float shift = bn2b[cc] - bn2m[cc] * scl;
#pragma unroll
        for (int t = 0; t < 2; ++t) {
#pragma unroll
            for (int i = 0; i < 8; ++i) {
                float z = (acc2[t][nt][i] + gadd) * scl + shift;
                z = fmaxf(z, 0.f);
                shw[(t * 16 + i + 8 * half) * SH_H16 + cc] = (_Float16)z;
            }
        }
    }
    __syncthreads();

    // ---- conv3: 2x (16x64) tiles, K=128 (A re-read from LDS f16) ----
    v8f acc3[2][4] = {};
    const _Float16* hrow0 = shw + l16 * SH_H16;
    const _Float16* hrow1 = shw + (16 + l16) * SH_H16;
#pragma unroll
    for (int c = 0; c < 4; ++c) {
        int kb = c * 32;
        int b0 = kb + 16 * half;
        int b1 = kb + 16 * half + 8;
        v8h p00 = *(const v8h*)(hrow0 + b0);
        v8h p01 = *(const v8h*)(hrow0 + b1);
        v8h p10 = *(const v8h*)(hrow1 + b0);
        v8h p11 = *(const v8h*)(hrow1 + b1);
        v16h a0, a1;
#pragma unroll
        for (int j = 0; j < 8; ++j) {
            a0[j] = p00[j];  a0[8 + j] = p01[j];
            a1[j] = p10[j];  a1[8 + j] = p11[j];
        }
        int kbB = kb + 16 * half;
#pragma unroll
        for (int nt = 0; nt < 4; ++nt) {
            v16h bb = *(const v16h*)(w3h + (nt * 16 + l16) * C2n + kbB);
            acc3[0][nt] = wmma_f16_32(a0, bb, acc3[0][nt]);
            acc3[1][nt] = wmma_f16_32(a1, bb, acc3[1][nt]);
        }
    }
    // bias + bn3 + relu -> h3 tiles (f32) in LDS
    float* sfw = shf + waveId * (2 * 16 * SH_F32);
#pragma unroll
    for (int nt = 0; nt < 4; ++nt) {
        int cc = nt * 16 + l16;
        float scl   = bn3g[cc] * rsqrtf(bn3v[cc] + EPSf);
        float shift = bn3b[cc] - bn3m[cc] * scl;
        float badd  = b3[cc];
#pragma unroll
        for (int t = 0; t < 2; ++t) {
#pragma unroll
            for (int i = 0; i < 8; ++i) {
                float z = (acc3[t][nt][i] + badd) * scl + shift;
                z = fmaxf(z, 0.f);
                sfw[(t * 16 + i + 8 * half) * SH_F32 + cc] = z;
            }
        }
    }
    __syncthreads();

    // ---- conv4: 2 channels; lane = (c = half, point = l16), 2 m-tiles ----
#pragma unroll
    for (int t = 0; t < 2; ++t) {
        int c = half;
        float val = b4[c];
#pragma unroll
        for (int k = 0; k < C3n; ++k)
            val += sfw[(t * 16 + l16) * SH_F32 + k] * w4[c * C3n + k];
        out[(size_t)bIdx * (NCn * Nn) + (size_t)c * Nn + (n0 + t * 16 + l16)] = val;
    }
}

// ---------------------------------------------------------------------------
extern "C" void kernel_launch(void* const* d_in, const int* in_sizes, int n_in,
                              void* d_out, int out_size, void* d_ws, size_t ws_size,
                              hipStream_t stream) {
    const float* gl_feats   = (const float*)d_in[0];
    const float* lo_feats   = (const float*)d_in[1];
    const float* centroids  = (const float*)d_in[2];
    /* d_in[3] np_cluster (int32): uniformly P=256 per the setup -> n>>8 */
    const float* fc1_w      = (const float*)d_in[4];
    const float* fc1_b      = (const float*)d_in[5];
    const float* fc2_w      = (const float*)d_in[6];
    const float* fc2_b      = (const float*)d_in[7];
    const float* in_proj_w  = (const float*)d_in[8];
    const float* in_proj_b  = (const float*)d_in[9];
    const float* out_proj_w = (const float*)d_in[10];
    const float* out_proj_b = (const float*)d_in[11];
    const float* conv2_w    = (const float*)d_in[12];
    const float* conv2_b    = (const float*)d_in[13];
    const float* bn2_gamma  = (const float*)d_in[14];
    const float* bn2_beta   = (const float*)d_in[15];
    const float* bn2_mean   = (const float*)d_in[16];
    const float* bn2_var    = (const float*)d_in[17];
    const float* conv3_w    = (const float*)d_in[18];
    const float* conv3_b    = (const float*)d_in[19];
    const float* bn3_gamma  = (const float*)d_in[20];
    const float* bn3_beta   = (const float*)d_in[21];
    const float* bn3_mean   = (const float*)d_in[22];
    const float* bn3_var    = (const float*)d_in[23];
    const float* conv4_w    = (const float*)d_in[24];
    const float* conv4_b    = (const float*)d_in[25];
    float* out = (float*)d_out;

    // workspace carve-up
    float* ws   = (float*)d_ws;
    float* x    = ws;                      // 262144 f32
    float* qkv  = x    + Bn * Sn * En;     // 786432 f32
    float* ctx  = qkv  + Bn * Sn * 3 * En; // 262144 f32
    float* attn = ctx  + Bn * Sn * En;     // 262144 f32
    float* g2   = attn + Bn * Sn * En;     // 131072 f32
    _Float16* w2h = (_Float16*)(g2 + Bn * Sn * C2n);  // 16384 f16 (32B aligned)
    _Float16* w3h = w2h + C2n * LDn;                  // 8192 f16

    // 0) weight f32->f16 conversion for the hot path
    convert_weights_kernel<<<(C2n * LDn + 255) / 256, 256, 0, stream>>>(
        conv2_w, conv3_w, w2h, w3h);

    // 1) positional MLP + transpose-add
    pos_x_kernel<<<(Bn * Sn * En + 255) / 256, 256, 0, stream>>>(
        gl_feats, centroids, fc1_w, fc1_b, fc2_w, fc2_b, x);

    // 2) qkv = x @ in_proj_w.T + b   (1024 x 768, K=256)
    gemm_wmma_f32<<<(64 * 48) / 4, 128, 0, stream>>>(
        x, En, in_proj_w, En, in_proj_b, qkv, 3 * En, Bn * Sn, 3 * En, En);

    // 3) attention -> ctx
    attention_kernel<<<(Bn * Hn * Sn) / 8, 256, 0, stream>>>(qkv, ctx);

    // 4) attn_out = ctx @ out_proj_w.T + b  (1024 x 256)
    gemm_wmma_f32<<<(64 * 16) / 4, 128, 0, stream>>>(
        ctx, En, out_proj_w, En, out_proj_b, attn, En, Bn * Sn, En, En);

    // 5) g2 = attn_out @ conv2_w[:,128:].T + conv2_b  (1024 x 128)
    gemm_wmma_f32<<<(64 * 8) / 4, 128, 0, stream>>>(
        attn, En, conv2_w + LDn, LDn + En, conv2_b, g2, C2n, Bn * Sn, C2n, En);

    // 6) fused point MLP (f16 WMMA): 262144 rows, 128 rows/block -> 2048 blocks
    fused_mlp_kernel<<<(Bn * Nn) / 128, 128, 0, stream>>>(
        lo_feats, g2, w2h,
        bn2_gamma, bn2_beta, bn2_mean, bn2_var,
        w3h, conv3_b,
        bn3_gamma, bn3_beta, bn3_mean, bn3_var,
        conv4_w, conv4_b, out);
}
